// BiLSTM_CRF_9534827397397
// MI455X (gfx1250) — matrix operand order repaired and checked by reading, not compiled, not run on hardware
//
#include <hip/hip_runtime.h>
#include <hip/hip_bf16.h>

// ---------------------------------------------------------------------------
// BiLSTM-CRF on gfx1250 (MI455X): bf16 WMMA for all matmuls, f32 accumulate.
//   V=50000 E=256 H2=512 H=256 K=12 B=32 L=512
// ---------------------------------------------------------------------------
#define E_    256
#define H2_   512
#define H_    256
#define K_    12
#define B_    32
#define L_    512
#define M_    (L_ * B_)      // 16384 rows (m = t*B + b)
#define G4H_  1024           // 4*H gate width
#define START_ 10
#define STOP_  11

typedef __attribute__((ext_vector_type(16))) __bf16 v16bf;
typedef __attribute__((ext_vector_type(8)))  float  v8f;

union Frag {
    v16bf v;
    uint4 q[2];
};

// gfx1250 async memory->LDS path (ASYNCcnt-tracked). Builtin signature
// (probe-confirmed from the round-2 diagnostic):
//   void __builtin_amdgcn_global_load_async_to_lds_b128(
//       int __vector(4) addrspace(1)* gsrc, int __vector(4) addrspace(3)* ldst,
//       int imm_offset, int cpol)
#if defined(__gfx1250__) && __has_builtin(__builtin_amdgcn_global_load_async_to_lds_b128) && __has_builtin(__builtin_amdgcn_s_wait_asynccnt)
#define USE_ASYNC_LDS 1
typedef int async_i4 __attribute__((vector_size(16)));
typedef __attribute__((address_space(1))) async_i4* gas_i4;
typedef __attribute__((address_space(3))) async_i4* las_i4;
#endif

__device__ __forceinline__ unsigned short f2bf(float f) {
    union { float f; unsigned int u; } x; x.f = f;
    unsigned int u = x.u;
    unsigned int r = 0x7FFFu + ((u >> 16) & 1u);   // round-to-nearest-even
    return (unsigned short)((u + r) >> 16);
}

// gfx1250 has a hardware v_tanh_f32 TRANS op; route tanh AND sigmoid through it.
#if defined(__gfx1250__) && __has_builtin(__builtin_amdgcn_tanhf)
__device__ __forceinline__ float fast_tanh(float x) { return __builtin_amdgcn_tanhf(x); }
#else
__device__ __forceinline__ float fast_tanh(float x) { return tanhf(x); }
#endif
__device__ __forceinline__ float sigm(float x) { return 0.5f * fast_tanh(0.5f * x) + 0.5f; }

// ---------------------------------------------------------------------------
// Embedding gather: tokens[B][L] -> xbf[m=t*B+b][E] (bf16)
// ---------------------------------------------------------------------------
__global__ __launch_bounds__(256) void k_gather(const int* __restrict__ tok,
                                                const float* __restrict__ emb,
                                                unsigned short* __restrict__ xbf) {
    int i = blockIdx.x * 256 + threadIdx.x;      // over M_*E_
    int m = i >> 8, e = i & 255;
    int t = m >> 5, b = m & 31;
    int tk = tok[b * L_ + t];
    xbf[i] = f2bf(emb[(size_t)tk * E_ + e]);
}

// f32 -> bf16 weight convert
__global__ __launch_bounds__(256) void k_cvt(const float* __restrict__ s,
                                             unsigned short* __restrict__ d, int n) {
    int i = blockIdx.x * 256 + threadIdx.x;
    if (i < n) d[i] = f2bf(s[i]);
}

// fc_w [12][512] -> padded [16][512] bf16 (zero rows 12..15)
__global__ __launch_bounds__(256) void k_cvt_fc(const float* __restrict__ s,
                                                unsigned short* __restrict__ d) {
    int i = blockIdx.x * 256 + threadIdx.x;      // 16*512
    int r = i >> 9, c = i & 511;
    d[i] = (r < K_) ? f2bf(s[r * H2_ + c]) : (unsigned short)0;
}

// ---------------------------------------------------------------------------
// Batched input projection: G[m][n] = sum_k A[m][k]*W[n][k] + bias[n]
// A bf16 [M][Kd] row-major, W bf16 [1024][Kd] row-major, G f32 [M][1024].
// Block tile 64(M) x 128(N), 8 waves as 4x2, each wave 16x64 via 4 WMMA tiles.
// Tiles staged to LDS with GLOBAL_LOAD_ASYNC_TO_LDS_B128 when available.
// ---------------------------------------------------------------------------
__global__ __launch_bounds__(256) void k_gemm(const unsigned short* __restrict__ A,
                                              const unsigned short* __restrict__ W,
                                              const float* __restrict__ bias,
                                              float* __restrict__ Gm, int Kd) {
    __shared__ unsigned short As[64 * 32];    // 4 KB
    __shared__ unsigned short Ws[128 * 32];   // 8 KB
    const int tid  = threadIdx.x;
    const int lane = tid & 31;
    const int w    = tid >> 5;
    const int m0   = blockIdx.x * 64;
    const int n0   = blockIdx.y * 128;
    const int Mt   = w >> 1;                  // 0..3
    const int nsub = (w & 1) * 64;
    const bool hi  = lane >= 16;
    const int  lr  = lane & 15;

    const v8f vzero = {0, 0, 0, 0, 0, 0, 0, 0};
    v8f acc[4];
#pragma unroll
    for (int j = 0; j < 4; ++j) acc[j] = vzero;

    const int rowA = tid >> 2, c4 = tid & 3;  // A: 256 16-B chunks
    for (int k0 = 0; k0 < Kd; k0 += 32) {
        __syncthreads();
#if defined(USE_ASYNC_LDS)
        {   // async memory->LDS DMA, 16 B per lane per op
            __builtin_amdgcn_global_load_async_to_lds_b128(
                (gas_i4)&A[(size_t)(m0 + rowA) * Kd + k0 + c4 * 8],
                (las_i4)&As[rowA * 32 + c4 * 8], 0, 0);
            int idx = tid;
#pragma unroll
            for (int rep = 0; rep < 2; ++rep) {
                int rw = idx >> 2, cw = idx & 3;
                __builtin_amdgcn_global_load_async_to_lds_b128(
                    (gas_i4)&W[(size_t)(n0 + rw) * Kd + k0 + cw * 8],
                    (las_i4)&Ws[rw * 32 + cw * 8], 0, 0);
                idx += 256;
            }
            __builtin_amdgcn_s_wait_asynccnt(0);
        }
#else
        {   // stage A: 64 rows x 32 bf16 = 256 x 16B chunks
            *(uint4*)&As[rowA * 32 + c4 * 8] =
                *(const uint4*)&A[(size_t)(m0 + rowA) * Kd + k0 + c4 * 8];
            // stage W: 128 rows -> 512 chunks, 2 per thread
            int idx = tid;
#pragma unroll
            for (int rep = 0; rep < 2; ++rep) {
                int rw = idx >> 2, cw = idx & 3;
                *(uint4*)&Ws[rw * 32 + cw * 8] =
                    *(const uint4*)&W[(size_t)(n0 + rw) * Kd + k0 + cw * 8];
                idx += 256;
            }
        }
#endif
        __syncthreads();
        Frag fa;
        {   // A fragment (16x32 bf16 layout): lanes 0-15 K0..7/16..23, lanes16-31 K8..15/24..31
            int r  = Mt * 16 + lr;
            int lo = r * 16 + (hi ? 4 : 0);          // dword index, 16 dwords/row
            fa.q[0] = *(const uint4*)&As[lo * 2];
            fa.q[1] = *(const uint4*)&As[(lo + 8) * 2];
        }
#pragma unroll
        for (int jt = 0; jt < 4; ++jt) {
            Frag fb;                                  // B fragment: col n = row n of W
            int n  = nsub + jt * 16 + lr;
            int lo = n * 16 + (hi ? 4 : 0);
            fb.q[0] = *(const uint4*)&Ws[lo * 2];
            fb.q[1] = *(const uint4*)&Ws[(lo + 8) * 2];
            acc[jt] = __builtin_amdgcn_wmma_f32_16x16x32_bf16(
                false, fa.v, false, fb.v, (short)0, acc[jt], false, false);
        }
    }
    // epilogue: C layout -> lane holds col (lane&15), rows Mt*16 + r + (hi?8:0)
    const int mrb = m0 + Mt * 16 + (hi ? 8 : 0);
#pragma unroll
    for (int jt = 0; jt < 4; ++jt) {
        int n = n0 + nsub + jt * 16 + lr;
        float bs = bias[n];
#pragma unroll
        for (int r = 0; r < 8; ++r)
            Gm[(size_t)(mrb + r) * G4H_ + n] = acc[jt][r] + bs;
    }
}

// ---------------------------------------------------------------------------
// Persistent recurrent scan, one block per direction.
// gates[32][1024] = G[t] (precomputed x@Wih^T + bias) + h@Whh^T, then LSTM cell.
// Wave w owns hidden columns j in [w*32, w*32+32) for ALL four gates, so the
// cell update is register-local. h (bf16) lives in LDS; c (f32) in registers.
// ---------------------------------------------------------------------------
__global__ __launch_bounds__(256) void k_rec(const float* __restrict__ Gall,
                                             const unsigned short* __restrict__ WhhAll,
                                             unsigned short* __restrict__ hout) {
    __shared__ unsigned short hsh[32 * 256];   // 16 KB, h[b][j] bf16
    const int dir = blockIdx.x;
    const float* Gm = Gall + (size_t)dir * M_ * G4H_;
    const unsigned short* Whh = WhhAll + (size_t)dir * G4H_ * H_;
    const int hoff = dir * H_;
    const int tid = threadIdx.x, lane = tid & 31, w = tid >> 5;
    const bool hi = lane >= 16;
    const int  lr = lane & 15;
    const int  jb = w * 32;

    for (int i = tid; i < 32 * 256; i += 256) hsh[i] = 0;
    float c[2][2][8];
#pragma unroll
    for (int Mt = 0; Mt < 2; ++Mt)
#pragma unroll
        for (int jt = 0; jt < 2; ++jt)
#pragma unroll
            for (int r = 0; r < 8; ++r) c[Mt][jt][r] = 0.0f;
    __syncthreads();

    for (int s = 0; s < L_; ++s) {
        const int t = dir ? (L_ - 1 - s) : s;
        v8f acc[4][2][2];
        // seed accumulators with the precomputed input projection + bias
#pragma unroll
        for (int q = 0; q < 4; ++q)
#pragma unroll
            for (int Mt = 0; Mt < 2; ++Mt)
#pragma unroll
                for (int jt = 0; jt < 2; ++jt) {
                    int n  = q * 256 + jb + jt * 16 + lr;
                    int mb = t * 32 + Mt * 16 + (hi ? 8 : 0);
#pragma unroll
                    for (int r = 0; r < 8; ++r)
                        acc[q][Mt][jt][r] = Gm[(size_t)(mb + r) * G4H_ + n];
                }
        // h @ Whh^T : M=32, N=1024, K=256 -> 8 k-steps of 16x16x32 bf16 WMMA
#pragma unroll
        for (int kk = 0; kk < 8; ++kk) {
            Frag fa[2];
#pragma unroll
            for (int Mt = 0; Mt < 2; ++Mt) {
                int b    = Mt * 16 + lr;
                int base = b * 128 + kk * 16;          // 128 dwords per h row
                int lo   = base + (hi ? 4 : 0);
                fa[Mt].q[0] = *(const uint4*)&hsh[lo * 2];
                fa[Mt].q[1] = *(const uint4*)&hsh[(lo + 8) * 2];
            }
#pragma unroll
            for (int q = 0; q < 4; ++q)
#pragma unroll
                for (int jt = 0; jt < 2; ++jt) {
                    Frag fb;                            // Whh row n = B column n (L2-resident)
                    int n  = q * 256 + jb + jt * 16 + lr;
                    int lo = n * 128 + kk * 16 + (hi ? 4 : 0);
                    fb.q[0] = *(const uint4*)&Whh[lo * 2];
                    fb.q[1] = *(const uint4*)&Whh[(lo + 8) * 2];
#pragma unroll
                    for (int Mt = 0; Mt < 2; ++Mt)
                        acc[q][Mt][jt] = __builtin_amdgcn_wmma_f32_16x16x32_bf16(
                            false, fa[Mt].v, false, fb.v, (short)0,
                            acc[q][Mt][jt], false, false);
                }
        }
        __syncthreads();    // all waves finished reading old h
        // LSTM cell (i,f,g,o quadruple is wave/lane-local by construction)
#pragma unroll
        for (int Mt = 0; Mt < 2; ++Mt)
#pragma unroll
            for (int jt = 0; jt < 2; ++jt) {
                int j     = jb + jt * 16 + lr;
                int bbase = Mt * 16 + (hi ? 8 : 0);
#pragma unroll
                for (int r = 0; r < 8; ++r) {
                    float ig = acc[0][Mt][jt][r], fg = acc[1][Mt][jt][r];
                    float gg = acc[2][Mt][jt][r], og = acc[3][Mt][jt][r];
                    float cc = sigm(fg) * c[Mt][jt][r] + sigm(ig) * fast_tanh(gg);
                    c[Mt][jt][r] = cc;
                    float hh = sigm(og) * fast_tanh(cc);
                    unsigned short hb = f2bf(hh);
                    int b = bbase + r;
                    hsh[b * 256 + j] = hb;
                    hout[(size_t)(t * 32 + b) * H2_ + hoff + j] = hb;
                }
            }
        __syncthreads();    // new h visible before next step's fragment loads
    }
}

// ---------------------------------------------------------------------------
// FC head: feats[m][n] = H[m][:] . fc_w[n][:] + fc_b[n], N padded to 16.
// One wave per 16-row M-tile, K=512 -> 16 WMMA.
// ---------------------------------------------------------------------------
__global__ __launch_bounds__(256) void k_fc(const unsigned short* __restrict__ Hin,
                                            const unsigned short* __restrict__ Wfc,
                                            const float* __restrict__ bfc,
                                            float* __restrict__ feats) {
    const int tid = threadIdx.x, lane = tid & 31, w = tid >> 5;
    const bool hi = lane >= 16;
    const int  lr = lane & 15;
    const int  m0 = blockIdx.x * 128 + w * 16;
    v8f acc = {0, 0, 0, 0, 0, 0, 0, 0};
#pragma unroll
    for (int kk = 0; kk < 16; ++kk) {
        Frag fa, fb;
        {
            int m    = m0 + lr;
            int base = m * 256 + kk * 16;      // 256 dwords per 512-wide bf16 row
            int lo   = base + (hi ? 4 : 0);
            fa.q[0] = *(const uint4*)&Hin[lo * 2];
            fa.q[1] = *(const uint4*)&Hin[(lo + 8) * 2];
        }
        {
            int lo = lr * 256 + kk * 16 + (hi ? 4 : 0);
            fb.q[0] = *(const uint4*)&Wfc[lo * 2];
            fb.q[1] = *(const uint4*)&Wfc[(lo + 8) * 2];
        }
        acc = __builtin_amdgcn_wmma_f32_16x16x32_bf16(
            false, fa.v, false, fb.v, (short)0, acc, false, false);
    }
    if (lr < K_) {
        float bs = bfc[lr];
        int mb = m0 + (hi ? 8 : 0);
#pragma unroll
        for (int r = 0; r < 8; ++r)
            feats[(size_t)(mb + r) * K_ + lr] = acc[r] + bs;
    }
}

// ---------------------------------------------------------------------------
// Viterbi over K=12 lattice: one block, 384 threads = (b,k) pairs.
// ---------------------------------------------------------------------------
__global__ __launch_bounds__(384) void k_vit(const float* __restrict__ feats,
                                             const float* __restrict__ trans,
                                             int* __restrict__ bp,
                                             int* __restrict__ out) {
    __shared__ float sc[32 * 12];
    __shared__ float tr[12 * 12];
    __shared__ float ns[32 * 12];
    const int tid = threadIdx.x;
    if (tid < 144) tr[tid] = trans[tid];
    const int b = tid / 12, k = tid % 12;
    __syncthreads();
    sc[tid] = feats[b * 12 + k] + tr[START_ * 12 + k];
    __syncthreads();
    for (int t = 1; t < L_; ++t) {
        float best = -3.0e38f; int bi = 0;
#pragma unroll
        for (int p = 0; p < 12; ++p) {
            float v = sc[b * 12 + p] + tr[p * 12 + k];
            if (v > best) { best = v; bi = p; }
        }
        bp[(size_t)(t - 1) * 384 + tid] = bi;
        float nv = best + feats[(size_t)t * 384 + tid];
        __syncthreads();
        ns[tid] = nv;
        __syncthreads();
        sc[tid] = ns[tid];
        __syncthreads();
    }
    sc[tid] += tr[k * 12 + STOP_];
    __syncthreads();
    if (tid < 32) {
        int bb = tid;
        float best = -3.0e38f; int bi = 0;
        for (int p = 0; p < 12; ++p) {
            float v = sc[bb * 12 + p];
            if (v > best) { best = v; bi = p; }
        }
        out[bb * L_ + (L_ - 1)] = bi;
        int st = bi;
        for (int t = L_ - 2; t >= 0; --t) {
            st = bp[(size_t)t * 384 + bb * 12 + st];
            out[bb * L_ + t] = st;
        }
    }
}

// ---------------------------------------------------------------------------
extern "C" void kernel_launch(void* const* d_in, const int* in_sizes, int n_in,
                              void* d_out, int out_size, void* d_ws, size_t ws_size,
                              hipStream_t stream) {
    const int*   tokens = (const int*)d_in[0];
    const float* embed  = (const float*)d_in[1];
    const float* w_ih0  = (const float*)d_in[2];
    const float* w_hh0  = (const float*)d_in[3];
    const float* b0     = (const float*)d_in[4];
    const float* w_ih   = (const float*)d_in[5];
    const float* w_hh   = (const float*)d_in[6];
    const float* bL     = (const float*)d_in[7];
    const float* fc_w   = (const float*)d_in[8];
    const float* fc_b   = (const float*)d_in[9];
    const float* trans  = (const float*)d_in[10];
    int* out = (int*)d_out;

    char* p = (char*)d_ws;
    auto carve = [&](size_t bytes) -> void* {
        void* r = (void*)p;
        p += (bytes + 255) & ~(size_t)255;
        return r;
    };
    unsigned short* xbf   = (unsigned short*)carve((size_t)M_ * E_ * 2);
    unsigned short* hA    = (unsigned short*)carve((size_t)M_ * H2_ * 2);
    unsigned short* hB    = (unsigned short*)carve((size_t)M_ * H2_ * 2);
    float*          Gbuf  = (float*)carve((size_t)2 * M_ * G4H_ * 4);
    unsigned short* wih0b = (unsigned short*)carve((size_t)2 * 1024 * 256 * 2);
    unsigned short* whh0b = (unsigned short*)carve((size_t)2 * 1024 * 256 * 2);
    unsigned short* wihb  = (unsigned short*)carve((size_t)6 * 1024 * 512 * 2);
    unsigned short* whhb  = (unsigned short*)carve((size_t)6 * 1024 * 256 * 2);
    unsigned short* fcwb  = (unsigned short*)carve((size_t)16 * 512 * 2);
    float*          feats = (float*)carve((size_t)M_ * K_ * 4);
    int*            bpbuf = (int*)carve((size_t)(L_ - 1) * 384 * 4);

    // embedding + weight conversion to bf16
    k_gather<<<(M_ * E_) / 256, 256, 0, stream>>>(tokens, embed, xbf);
    k_cvt<<<(2 * 1024 * 256) / 256, 256, 0, stream>>>(w_ih0, wih0b, 2 * 1024 * 256);
    k_cvt<<<(2 * 1024 * 256) / 256, 256, 0, stream>>>(w_hh0, whh0b, 2 * 1024 * 256);
    k_cvt<<<(6 * 1024 * 512) / 256, 256, 0, stream>>>(w_ih, wihb, 6 * 1024 * 512);
    k_cvt<<<(6 * 1024 * 256) / 256, 256, 0, stream>>>(w_hh, whhb, 6 * 1024 * 256);
    k_cvt_fc<<<(16 * 512) / 256, 256, 0, stream>>>(fc_w, fcwb);

    dim3 ggrid(M_ / 64, G4H_ / 128);

    // layer 0: E=256 input
    for (int d = 0; d < 2; ++d)
        k_gemm<<<ggrid, 256, 0, stream>>>(xbf, wih0b + (size_t)d * 1024 * 256,
                                          b0 + (size_t)d * 1024,
                                          Gbuf + (size_t)d * M_ * G4H_, 256);
    k_rec<<<2, 256, 0, stream>>>(Gbuf, whh0b, hA);

    unsigned short* hin = hA;
    unsigned short* hnx = hB;
    for (int l = 0; l < 3; ++l) {
        for (int d = 0; d < 2; ++d)
            k_gemm<<<ggrid, 256, 0, stream>>>(hin, wihb + (size_t)(l * 2 + d) * 1024 * 512,
                                              bL + (size_t)(l * 2 + d) * 1024,
                                              Gbuf + (size_t)d * M_ * G4H_, 512);
        k_rec<<<2, 256, 0, stream>>>(Gbuf, whhb + (size_t)(l * 2) * 1024 * 256, hnx);
        unsigned short* tmp = hin; hin = hnx; hnx = tmp;
    }

    k_fc<<<M_ / 128, 256, 0, stream>>>(hin, fcwb, fc_b, feats);
    k_vit<<<1, 384, 0, stream>>>(feats, trans, bpbuf, out);
}
